// LinearAttentionExpert_48627619725843
// MI455X (gfx1250) — matrix operand order repaired and checked by reading, not compiled, not run on hardware
//
#include <hip/hip_runtime.h>

// ---------------------------------------------------------------------------
// LinearAttentionExpert pipeline for MI455X (gfx1250, wave32, WMMA).
// B=4, N=8192, H=E=1024.  All GEMMs run as bf16 WMMA (16x16x32) w/ f32 accum.
// GEMM: 128x128 block tile, BK=64, 8 waves, 16 WMMA/iter, double-buffered LDS:
// tile t+1 streams in via GLOBAL_LOAD_ASYNC_TO_LDS_B128 / ds stores while the
// WMMAs for tile t execute; one s_wait_asynccnt + one barrier per iteration.
// ---------------------------------------------------------------------------

typedef __attribute__((ext_vector_type(16))) __bf16         v16bf;
typedef __attribute__((ext_vector_type(8)))  float          v8f;
typedef __attribute__((ext_vector_type(8)))  unsigned short v8u16;
typedef int v4i __attribute__((vector_size(16)));   // matches builtin param type

#define BATCH  4
#define NSEQ   8192
#define EDIM   1024
#define ROWS   (BATCH * NSEQ)          // 32768

#if __has_builtin(__builtin_amdgcn_global_load_async_to_lds_b128)
#define ASYNC_LDS 1
#define GPTR(p) ((__attribute__((address_space(1))) v4i*)(p))
#define LPTR(p) ((__attribute__((address_space(3))) v4i*)(p))
#else
#define ASYNC_LDS 0
#endif

__device__ __forceinline__ unsigned short f2bf(float f) {
    unsigned int u = __builtin_bit_cast(unsigned int, f);
    u += 0x7FFFu + ((u >> 16) & 1u);   // round-to-nearest-even
    return (unsigned short)(u >> 16);
}
__device__ __forceinline__ float bf2f(unsigned short h) {
    unsigned int u = ((unsigned int)h) << 16;
    return __builtin_bit_cast(float, u);
}
// pack 4 bf16 (consecutive fragment elements) into one ds_store_b64
__device__ __forceinline__ void pack_store_b64(unsigned short* dst,
                                               unsigned short a0, unsigned short a1,
                                               unsigned short a2, unsigned short a3) {
    uint2 v;
    v.x = (unsigned)a0 | ((unsigned)a1 << 16);
    v.y = (unsigned)a2 | ((unsigned)a3 << 16);
    *(uint2*)dst = v;
}

// ---------------------------------------------------------------- converts
__global__ void cvt_f32_bf16(const float* __restrict__ src,
                             unsigned short* __restrict__ dst, long long n) {
    long long i = (long long)blockIdx.x * blockDim.x + threadIdx.x;
    long long stride = (long long)gridDim.x * blockDim.x;
    for (; i < n; i += stride) dst[i] = f2bf(src[i]);
}

// ---------------------------------------------------------------- ksum (deterministic two-stage)
__global__ void ksum_partial(const unsigned short* __restrict__ qkv,
                             float* __restrict__ part) {
    int e = blockIdx.x * 256 + threadIdx.x;
    int b = blockIdx.y;
    int c = blockIdx.z;                         // chunk 0..31 (256 rows each)
    const unsigned short* kp = qkv + (size_t)b * NSEQ * (3 * EDIM)
                                   + (size_t)(c * 256) * (3 * EDIM) + EDIM + e;
    float s = 0.f;
    for (int n = 0; n < 256; ++n) s += bf2f(kp[(size_t)n * (3 * EDIM)]);
    part[((size_t)c * BATCH + b) * EDIM + e] = s;
}
__global__ void ksum_reduce(const float* __restrict__ part,
                            float* __restrict__ ksum) {
    int e = blockIdx.x * 256 + threadIdx.x;
    int b = blockIdx.y;
    float s = 0.f;
    for (int c = 0; c < 32; ++c) s += part[((size_t)c * BATCH + b) * EDIM + e];
    ksum[(size_t)b * EDIM + e] = s;
}

// ---------------------------------------------------------------- z = 1/(q.ksum + eps)
__global__ void zcalc(const unsigned short* __restrict__ qkv,
                      const float* __restrict__ ksum,
                      float* __restrict__ z) {
    int lane = threadIdx.x & 31;
    int wid  = threadIdx.x >> 5;
    int row  = blockIdx.x * 8 + wid;
    int b    = row / NSEQ;
    const unsigned short* q = qkv + (size_t)row * (3 * EDIM);
    const float* ks = ksum + (size_t)b * EDIM;
    float s = 0.f;
    for (int e = lane; e < EDIM; e += 32) s += bf2f(q[e]) * ks[e];
    #pragma unroll
    for (int off = 16; off; off >>= 1) s += __shfl_xor(s, off, 32);
    if (lane == 0) z[row] = 1.0f / (s + 1e-6f);
}

// ---------------------------------------------------------------- SwiGLU elementwise
__global__ void swiglu(const unsigned short* __restrict__ h,
                       unsigned short* __restrict__ g) {
    long long total = (long long)ROWS * EDIM;
    long long i = (long long)blockIdx.x * blockDim.x + threadIdx.x;
    long long stride = (long long)gridDim.x * blockDim.x;
    for (; i < total; i += stride) {
        long long row = i >> 10;
        int col = (int)(i & (EDIM - 1));
        float a  = bf2f(h[row * (2 * EDIM) + col]);
        float gt = bf2f(h[row * (2 * EDIM) + EDIM + col]);
        float sg = gt / (1.0f + __expf(-gt));   // silu
        g[i] = f2bf(a * sg);
    }
}

// ---------------------------------------------------------------- bf16 WMMA GEMM
// LDS tiles in WMMA fragment-major layout: [stage][kt][subtile][lane][16].
// A frag element map: lane = mrel + 16*((k>>3)&1), e = (k&7)|((k&16)>>1)
// B frag element map: lane = nrel + 16*((k>>3)&1), e = (k&7)|((k&16)>>1)
template <bool TRANS_A, bool OUT_F32>
__global__ __launch_bounds__(256)
void gemm_bf16_wmma(const unsigned short* __restrict__ A, int lda, long long strideA,
                    const unsigned short* __restrict__ B, int ldb, long long strideB,
                    void* __restrict__ Cv, int ldc, long long strideC,
                    const float* __restrict__ bias,
                    const float* __restrict__ zscale, long long strideZ,
                    int M, int N, int K) {
    __shared__ alignas(32) unsigned short As[2 * 2 * 8 * 32 * 16];   // 32 KB
    __shared__ alignas(32) unsigned short Bs[2 * 2 * 8 * 32 * 16];   // 32 KB

    const int tid  = threadIdx.x;
    const int lane = tid & 31;
    const int wid  = tid >> 5;
    const int wmi  = wid >> 2;        // wave M index 0..1
    const int wni  = wid & 3;         // wave N index 0..3

    const int bm = blockIdx.y * 128;
    const int bn = blockIdx.x * 128;
    const int batch = blockIdx.z;

    const unsigned short* Ab = A + (size_t)batch * strideA;
    const unsigned short* Bb = B + (size_t)batch * strideB;

    // loader coords (fixed per thread)
    const int a_row = tid >> 1;               // !TRANS_A: m row 0..127
    const int a_kh  = (tid & 1) << 5;         // !TRANS_A: k half 0 or 32
    const int qk    = (tid >> 4) << 2;        // TRANS_A / B: k quad base 0..60
    const int qcol  = (tid & 15) << 3;        // TRANS_A / B: 8-wide col chunk

    v8f acc[4][2];
    #pragma unroll
    for (int i = 0; i < 4; ++i)
        #pragma unroll
        for (int j = 0; j < 2; ++j) acc[i][j] = (v8f)0.0f;

    // ---- stream one BK=64 tile pair into LDS stage `stage`
    auto load_tiles = [&](int k0, int stage) {
        unsigned short* AsS = &As[stage * 8192];
        unsigned short* BsS = &Bs[stage * 8192];
        // B tile: 4 rows of a K-quad x 8 cols -> one (2addr) b64 store per col
        v8u16 rb[4];
        #pragma unroll
        for (int r = 0; r < 4; ++r)
            rb[r] = *(const v8u16*)(Bb + (size_t)(k0 + qk + r) * ldb + bn + qcol);
        {
            const int kt = qk >> 5, krel = qk & 31;
            const int hf = (krel >> 3) & 1;
            const int e  = (krel & 7) | ((krel & 16) >> 1);
            #pragma unroll
            for (int jn = 0; jn < 8; ++jn) {
                int n = qcol + jn;
                pack_store_b64(&BsS[kt * 4096 + (n >> 4) * 512 + ((n & 15) + (hf << 4)) * 16 + e],
                               rb[0][jn], rb[1][jn], rb[2][jn], rb[3][jn]);
            }
        }
        if (TRANS_A) {
            v8u16 ra[4];
            #pragma unroll
            for (int r = 0; r < 4; ++r)
                ra[r] = *(const v8u16*)(Ab + (size_t)(k0 + qk + r) * lda + bm + qcol);
            const int kt = qk >> 5, krel = qk & 31;
            const int hf = (krel >> 3) & 1;
            const int e  = (krel & 7) | ((krel & 16) >> 1);
            #pragma unroll
            for (int jm = 0; jm < 8; ++jm) {
                int m = qcol + jm;
                pack_store_b64(&AsS[kt * 4096 + (m >> 4) * 512 + ((m & 15) + (hf << 4)) * 16 + e],
                               ra[0][jm], ra[1][jm], ra[2][jm], ra[3][jm]);
            }
        } else {
            // 8 consecutive K values = 16B = exactly one fragment chunk
            const int mt = a_row >> 4, mrel = a_row & 15;
            unsigned short* base = &AsS[(a_kh >> 5) * 4096 + mt * 512];
            const unsigned short* src = Ab + (size_t)(bm + a_row) * lda + k0 + a_kh;
#if ASYNC_LDS
            __builtin_amdgcn_global_load_async_to_lds_b128(
                GPTR(src + 0),  LPTR(&base[mrel * 16 + 0]), 0, 0);
            __builtin_amdgcn_global_load_async_to_lds_b128(
                GPTR(src + 8),  LPTR(&base[(mrel + 16) * 16 + 0]), 0, 0);
            __builtin_amdgcn_global_load_async_to_lds_b128(
                GPTR(src + 16), LPTR(&base[mrel * 16 + 8]), 0, 0);
            __builtin_amdgcn_global_load_async_to_lds_b128(
                GPTR(src + 24), LPTR(&base[(mrel + 16) * 16 + 8]), 0, 0);
#else
            v8u16 ra[4];
            #pragma unroll
            for (int g = 0; g < 4; ++g) ra[g] = *(const v8u16*)(src + 8 * g);
            *(v8u16*)&base[mrel * 16 + 0]        = ra[0];  // k 0..7   half0
            *(v8u16*)&base[(mrel + 16) * 16 + 0] = ra[1];  // k 8..15  half1
            *(v8u16*)&base[mrel * 16 + 8]        = ra[2];  // k16..23  half0
            *(v8u16*)&base[(mrel + 16) * 16 + 8] = ra[3];  // k24..31  half1
#endif
        }
    };

    auto wait_streams = [&]() {
#if ASYNC_LDS
        if (!TRANS_A) {
#if __has_builtin(__builtin_amdgcn_s_wait_asynccnt)
            __builtin_amdgcn_s_wait_asynccnt(0);
#else
            asm volatile("s_wait_asynccnt 0" ::: "memory");
#endif
        }
#endif
    };

    // ---- 2 K-subtiles x 8 WMMAs from LDS stage `stage`
    auto compute_stage = [&](int stage) {
        const unsigned short* AsS = &As[stage * 8192];
        const unsigned short* BsS = &Bs[stage * 8192];
        #pragma unroll
        for (int kt = 0; kt < 2; ++kt) {
            v16bf afrag[4];
            #pragma unroll
            for (int i = 0; i < 4; ++i)
                afrag[i] = *(const v16bf*)&AsS[kt * 4096 + (wmi * 4 + i) * 512 + lane * 16];
            v16bf bfrag[2];
            #pragma unroll
            for (int j = 0; j < 2; ++j)
                bfrag[j] = *(const v16bf*)&BsS[kt * 4096 + (wni * 2 + j) * 512 + lane * 16];

            #pragma unroll
            for (int i = 0; i < 4; ++i)
                #pragma unroll
                for (int j = 0; j < 2; ++j)
                    acc[i][j] = __builtin_amdgcn_wmma_f32_16x16x32_bf16(
                        false, afrag[i], false, bfrag[j],
                        (short)0, acc[i][j], false, false);
        }
    };

    // ---- software-pipelined main loop (double-buffered)
    const int ntile = K >> 6;
    load_tiles(0, 0);
    wait_streams();
    __syncthreads();
    for (int t = 0; t < ntile; ++t) {
        const int stage = t & 1;
        if (t + 1 < ntile) {
            load_tiles((t + 1) << 6, stage ^ 1);     // stream next tile (other stage)
            if (t + 2 < ntile) {                     // prefetch tile t+2
                const int kp = (t + 2) << 6;
                __builtin_prefetch((const void*)(Bb + (size_t)(kp + qk) * ldb + bn + qcol), 0, 3);
                if (TRANS_A)
                    __builtin_prefetch((const void*)(Ab + (size_t)(kp + qk) * lda + bm + qcol), 0, 3);
                else
                    __builtin_prefetch((const void*)(Ab + (size_t)(bm + a_row) * lda + kp + a_kh), 0, 3);
            }
        }
        compute_stage(stage);                        // overlaps with async stream-in
        wait_streams();
        __syncthreads();
    }

    // ---- epilogue: bias, optional per-row z-scale, bf16 or f32 store
    float* Cf = (float*)Cv;
    unsigned short* Ch = (unsigned short*)Cv;
    const float* zs = zscale ? (zscale + (size_t)batch * strideZ) : nullptr;
    const int half  = lane >> 4;
    const int col_l = lane & 15;

    #pragma unroll
    for (int i = 0; i < 4; ++i) {
        #pragma unroll
        for (int j = 0; j < 2; ++j) {
            int n = bn + (wni * 2 + j) * 16 + col_l;
            float bv = bias ? bias[n] : 0.0f;
            #pragma unroll
            for (int r = 0; r < 8; ++r) {
                int m = bm + (wmi * 4 + i) * 16 + (half << 3) + r;
                float val = acc[i][j][r] + bv;
                if (zs) val *= zs[m];
                size_t idx = (size_t)batch * strideC + (size_t)m * ldc + n;
                if (OUT_F32) Cf[idx] = val;
                else         Ch[idx] = f2bf(val);
            }
        }
    }
}

// ---------------------------------------------------------------------------
extern "C" void kernel_launch(void* const* d_in, const int* in_sizes, int n_in,
                              void* d_out, int out_size, void* d_ws, size_t ws_size,
                              hipStream_t stream) {
    (void)in_sizes; (void)n_in; (void)out_size; (void)ws_size;

    const float* x       = (const float*)d_in[0];
    const float* qkv_w   = (const float*)d_in[1];
    const float* qkv_b   = (const float*)d_in[2];
    const float* sw_w    = (const float*)d_in[3];
    const float* sw_w_b  = (const float*)d_in[4];
    const float* sw_v    = (const float*)d_in[5];
    const float* sw_v_b  = (const float*)d_in[6];
    const float* proj_w  = (const float*)d_in[7];
    const float* proj_b  = (const float*)d_in[8];
    float* out = (float*)d_out;

    char* ws = (char*)d_ws;
    size_t off = 0;
    auto take = [&](size_t bytes) -> char* {
        char* p = ws + off;
        off += (bytes + 255) & ~(size_t)255;
        return p;
    };
    unsigned short* x_bf   = (unsigned short*)take(sizeof(short) * (size_t)ROWS * EDIM);
    unsigned short* qkv_bf = (unsigned short*)take(sizeof(short) * (size_t)ROWS * 3 * EDIM);
    unsigned short* out_bf = (unsigned short*)take(sizeof(short) * (size_t)ROWS * EDIM);
    unsigned short* h_bf   = (unsigned short*)take(sizeof(short) * (size_t)ROWS * 2 * EDIM);
    unsigned short* g_bf   = (unsigned short*)take(sizeof(short) * (size_t)ROWS * EDIM);
    unsigned short* s_bf   = (unsigned short*)take(sizeof(short) * (size_t)ROWS * EDIM);
    unsigned short* kv_bf  = (unsigned short*)take(sizeof(short) * (size_t)BATCH * EDIM * EDIM);
    unsigned short* wqkv   = (unsigned short*)take(sizeof(short) * (size_t)EDIM * 3 * EDIM);
    unsigned short* wsw    = (unsigned short*)take(sizeof(short) * (size_t)EDIM * 2 * EDIM);
    unsigned short* wsv    = (unsigned short*)take(sizeof(short) * (size_t)EDIM * EDIM);
    unsigned short* wproj  = (unsigned short*)take(sizeof(short) * (size_t)EDIM * EDIM);
    float* ksum  = (float*)take(sizeof(float) * (size_t)BATCH * EDIM);
    float* kpart = (float*)take(sizeof(float) * 32 * (size_t)BATCH * EDIM);
    float* zbuf  = (float*)take(sizeof(float) * (size_t)ROWS);

    // 1) f32 -> bf16
    cvt_f32_bf16<<<8192, 256, 0, stream>>>(x,      x_bf,  (long long)ROWS * EDIM);
    cvt_f32_bf16<<<2048, 256, 0, stream>>>(qkv_w,  wqkv,  (long long)EDIM * 3 * EDIM);
    cvt_f32_bf16<<<2048, 256, 0, stream>>>(sw_w,   wsw,   (long long)EDIM * 2 * EDIM);
    cvt_f32_bf16<<<1024, 256, 0, stream>>>(sw_v,   wsv,   (long long)EDIM * EDIM);
    cvt_f32_bf16<<<1024, 256, 0, stream>>>(proj_w, wproj, (long long)EDIM * EDIM);

    // 2) qkv = x @ qkv_w + qkv_b
    gemm_bf16_wmma<false, false><<<dim3(3 * EDIM / 128, ROWS / 128, 1), 256, 0, stream>>>(
        x_bf, EDIM, 0, wqkv, 3 * EDIM, 0, qkv_bf, 3 * EDIM, 0,
        qkv_b, nullptr, 0, ROWS, 3 * EDIM, EDIM);

    // 3) ksum
    ksum_partial<<<dim3(EDIM / 256, BATCH, 32), 256, 0, stream>>>(qkv_bf, kpart);
    ksum_reduce<<<dim3(EDIM / 256, BATCH), 256, 0, stream>>>(kpart, ksum);

    // 4) kv[b] = k^T @ v  (M=N=1024, K=8192)
    gemm_bf16_wmma<true, false><<<dim3(EDIM / 128, EDIM / 128, BATCH), 256, 0, stream>>>(
        qkv_bf + EDIM,     3 * EDIM, (long long)NSEQ * 3 * EDIM,
        qkv_bf + 2 * EDIM, 3 * EDIM, (long long)NSEQ * 3 * EDIM,
        kv_bf, EDIM, (long long)EDIM * EDIM,
        nullptr, nullptr, 0, EDIM, EDIM, NSEQ);

    // 5) z = 1/(q.ksum + eps)
    zcalc<<<ROWS / 8, 256, 0, stream>>>(qkv_bf, ksum, zbuf);

    // 6) out = (q @ kv) * z
    gemm_bf16_wmma<false, false><<<dim3(EDIM / 128, NSEQ / 128, BATCH), 256, 0, stream>>>(
        qkv_bf, 3 * EDIM, (long long)NSEQ * 3 * EDIM,
        kv_bf, EDIM, (long long)EDIM * EDIM,
        out_bf, EDIM, (long long)NSEQ * EDIM,
        nullptr, zbuf, NSEQ, NSEQ, EDIM, EDIM);

    // 7) h = out @ sw_w + sw_w_b
    gemm_bf16_wmma<false, false><<<dim3(2 * EDIM / 128, ROWS / 128, 1), 256, 0, stream>>>(
        out_bf, EDIM, 0, wsw, 2 * EDIM, 0, h_bf, 2 * EDIM, 0,
        sw_w_b, nullptr, 0, ROWS, 2 * EDIM, EDIM);

    // 8) g = a * silu(gate)
    swiglu<<<8192, 256, 0, stream>>>(h_bf, g_bf);

    // 9) s = g @ sw_v + sw_v_b
    gemm_bf16_wmma<false, false><<<dim3(EDIM / 128, ROWS / 128, 1), 256, 0, stream>>>(
        g_bf, EDIM, 0, wsv, EDIM, 0, s_bf, EDIM, 0,
        sw_v_b, nullptr, 0, ROWS, EDIM, EDIM);

    // 10) final = s @ proj_w + proj_b -> f32 d_out
    gemm_bf16_wmma<false, true><<<dim3(EDIM / 128, ROWS / 128, 1), 256, 0, stream>>>(
        s_bf, EDIM, 0, wproj, EDIM, 0, out, EDIM, 0,
        proj_b, nullptr, 0, ROWS, EDIM, EDIM);
}
